// MyGraphUNet_62423054680394
// MI455X (gfx1250) — matrix operand order repaired and checked by reading, not compiled, not run on hardware
//
#include <hip/hip_runtime.h>
#include <hip/hip_bf16.h>

typedef __attribute__((ext_vector_type(16))) _Float16 v16h;
typedef __attribute__((ext_vector_type(8)))  _Float16 v8h_t;
typedef __attribute__((ext_vector_type(8)))  float    v8f;

#define NNODES 4096
#define CH     256
#define CIN    128
#define DEPTHN 3

static __host__ __device__ inline size_t cdiv_sz(size_t a, size_t b) { return (a + b - 1) / b; }

// ---------------------------------------------------------------------------
// WMMA GEMM:  C[M,N] = Ah[M,K] * B[K,N],  B supplied pre-transposed:
//   Ah : f16 row-major, leading dim lda (halves)
//   Bt : f16 row-major N x K (i.e. B^T), leading dim ldb (halves)
//   C  : f32 row-major, leading dim ldc
// Requirements (all satisfied here): M%256==0, N%64==0, K%32==0,
// lda/ldb multiples of 64 halves, buffers 256B aligned.
//
// Block = 256 threads = 8 waves. Block tile 256(M) x 64(N):
//   - B panel (64 x 32 per K-step) staged cooperatively in LDS,
//     double-buffered, shared by all 8 waves (8x B-traffic reduction).
//   - Each wave owns 32 rows: 2 A fragments x 4 B fragments -> 8 WMMAs
//     per K-step (2x A reuse, back-to-back v_wmma issue).
// Global traffic per block per K-step: 16KB A + 4KB B for 64 WMMAs
// (1.05 MFLOP) -> ~52 flop/byte: compute-bound on the f16 WMMA pipe.
// ---------------------------------------------------------------------------
__global__ __launch_bounds__(256) void k_wmma_gemm(
    const _Float16* __restrict__ Ah, int lda,
    const _Float16* __restrict__ Bt, int ldb,
    float* __restrict__ C, int ldc,
    int M, int N, int K)
{
  __shared__ __align__(32) _Float16 lb[2][64 * 32];   // 8KB double-buffered B panel

  const int lane = threadIdx.x & 31;
  const int wave = threadIdx.x >> 5;
  const int tileN0 = blockIdx.x * 64;
  const int tileM  = blockIdx.y * 256 + wave * 32;    // M%256==0: no guard needed
  (void)M;

  // cooperative B staging: thread t copies 8 halves of column (t>>2)
  const int colS = threadIdx.x >> 2;                  // 0..63
  const int segS = (threadIdx.x & 3) * 8;             // 0,8,16,24 halves
  const _Float16* bsrc = Bt + (size_t)(tileN0 + colS) * ldb + segS;

  v8h_t breg = *(const v8h_t*)(bsrc);                 // k0 = 0 slab
  *(v8h_t*)&lb[0][colS * 32 + segS] = breg;
  __syncthreads();

  const int rowA = tileM + (lane & 15);
  const int aSel = (lane >= 16) ? 16 : 0;             // A frag K half-select
  const int bSel = (lane >= 16) ? 16 : 0;             // B frag K half-select

  v8f acc[2][4] = {};
  int buf = 0;

  for (int k0 = 0; k0 < K; k0 += 32) {
    const bool more = (k0 + 32) < K;                  // uniform
    if (more) breg = *(const v8h_t*)(bsrc + k0 + 32); // prefetch next B slab to regs

    // A fragments for rows [tileM..+15] and [tileM+16..+31]:
    // lanes 0-15 hold K=k0..k0+15 in a[0..15];
    // lanes 16-31 hold K=k0+16..k0+31.
    union { v16h v; v8h_t h[2]; } af[2];
    const _Float16* ap0 = Ah + (size_t)rowA * lda + k0 + aSel;
    const _Float16* ap1 = ap0 + (size_t)16 * lda;
    af[0].h[0] = *(const v8h_t*)(ap0);
    af[0].h[1] = *(const v8h_t*)(ap0 + 8);
    af[1].h[0] = *(const v8h_t*)(ap1);
    af[1].h[1] = *(const v8h_t*)(ap1 + 8);
    if (more) {
      __builtin_prefetch(ap0 + 32, 0, 0);
      __builtin_prefetch(ap1 + 32, 0, 0);
    }

#pragma unroll
    for (int t = 0; t < 4; ++t) {
      // B fragment: lane holds column N = 16*t + (lane&15); lanes 0-15 hold
      // K=k0..k0+15, lanes 16-31 K=k0+16..k0+31 (contiguous in LDS slab).
      v16h b = *(const v16h*)&lb[buf][(16 * t + (lane & 15)) * 32 + bSel];
      acc[0][t] = __builtin_amdgcn_wmma_f32_16x16x32_f16(
          false, af[0].v, false, b, (short)0, acc[0][t], false, false);
      acc[1][t] = __builtin_amdgcn_wmma_f32_16x16x32_f16(
          false, af[1].v, false, b, (short)0, acc[1][t], false, false);
    }

    if (more) {
      *(v8h_t*)&lb[buf ^ 1][colS * 32 + segS] = breg;
      __syncthreads();
      buf ^= 1;
    }
  }

  // C/D layout: lane L, vgpr v -> (M = v + 8*(L>=16), N = L&15)
#pragma unroll
  for (int s = 0; s < 2; ++s) {
    const int r0 = tileM + 16 * s + ((lane >= 16) ? 8 : 0);
#pragma unroll
    for (int t = 0; t < 4; ++t) {
      float* cp = C + (size_t)r0 * ldc + tileN0 + 16 * t + (lane & 15);
#pragma unroll
      for (int v = 0; v < 8; ++v) cp[(size_t)v * ldc] = acc[s][t][v];
    }
  }
}

// ---------------------------------------------------------------------------
// Elementwise / graph kernels
// ---------------------------------------------------------------------------
__global__ void k_zero_f32(float* __restrict__ p, size_t n) {
  size_t i   = (size_t)blockIdx.x * blockDim.x + threadIdx.x;
  size_t str = (size_t)gridDim.x * blockDim.x;
  for (; i < n; i += str) p[i] = 0.f;
}

__global__ void k_build_adj(const int* __restrict__ src, const int* __restrict__ dst,
                            float* __restrict__ A, int E, int n) {
  int e = blockIdx.x * blockDim.x + threadIdx.x;
  if (e < E) atomicAdd(&A[(size_t)src[e] * n + dst[e]], 1.f);
}

// dinv[c] = rsqrt(colsum(Ahat)[c]), Ahat = A + diag(2 where diag==0)  (improved)
__global__ void k_dinv(const float* __restrict__ A, float* __restrict__ dinv, int n) {
  int c = blockIdx.x * blockDim.x + threadIdx.x;
  if (c >= n) return;
  float s = 0.f;
  for (int r = 0; r < n; ++r) s += A[(size_t)r * n + c];
  if (A[(size_t)c * n + c] == 0.f) s += 2.f;
  dinv[c] = rsqrtf(s);
}

__global__ void k_tohalf(const float* __restrict__ a, _Float16* __restrict__ h, size_t n) {
  size_t i = (size_t)blockIdx.x * blockDim.x + threadIdx.x;
  if (i < n) h[i] = (_Float16)a[i];
}

// dst[cols][rows] = src[rows][cols]^T, optional per-source-row scale
__global__ void k_transpose_half(const float* __restrict__ src, const float* __restrict__ scale,
                                 _Float16* __restrict__ dst, int rows, int cols) {
  size_t idx = (size_t)blockIdx.x * blockDim.x + threadIdx.x;
  if (idx >= (size_t)rows * cols) return;
  int i = (int)(idx / cols), j = (int)(idx % cols);
  float v = src[idx];
  if (scale) v *= scale[i];
  dst[(size_t)j * rows + i] = (_Float16)v;
}

// h = Ahat^T in f16: h[i*n+j] = A[j*n+i] (+2 on empty diagonal)
__global__ void k_ahatT_half(const float* __restrict__ A, _Float16* __restrict__ h, int n) {
  size_t idx = (size_t)blockIdx.x * blockDim.x + threadIdx.x;
  if (idx >= (size_t)n * n) return;
  int i = (int)(idx / n), j = (int)(idx % n);
  float v = A[(size_t)j * n + i];
  if (i == j && v == 0.f) v = 2.f;
  h[idx] = (_Float16)v;
}

// h = (A+I) or (A+I)^T in f16
__global__ void k_aug_half(const float* __restrict__ A, _Float16* __restrict__ h, int n, int transposed) {
  size_t idx = (size_t)blockIdx.x * blockDim.x + threadIdx.x;
  if (idx >= (size_t)n * n) return;
  int i = (int)(idx / n), j = (int)(idx % n);
  float v = transposed ? A[(size_t)j * n + i] : A[idx];
  if (i == j) v += 1.f;
  h[idx] = (_Float16)v;
}

__global__ void k_zero_diag(float* __restrict__ A, int n) {
  int i = blockIdx.x * blockDim.x + threadIdx.x;
  if (i < n) A[(size_t)i * n + i] = 0.f;
}

// x_out = (relu?)(dinv[i]*z + b[j])
__global__ void k_gcn_epi(const float* __restrict__ z, const float* __restrict__ dinv,
                          const float* __restrict__ b, float* __restrict__ xo,
                          int n, int c, int relu) {
  size_t idx = (size_t)blockIdx.x * blockDim.x + threadIdx.x;
  if (idx >= (size_t)n * c) return;
  int i = (int)(idx / c), j = (int)(idx % c);
  float v = dinv[i] * z[idx] + b[j];
  xo[idx] = relu ? fmaxf(v, 0.f) : v;
}

__global__ void k_copy_f32(const float* __restrict__ a, float* __restrict__ b, size_t n) {
  size_t i = (size_t)blockIdx.x * blockDim.x + threadIdx.x;
  if (i < n) b[i] = a[i];
}

__global__ void k_scatter_add(const int* __restrict__ perm, const float* __restrict__ xc,
                              float* __restrict__ xo, int k, int c) {
  size_t idx = (size_t)blockIdx.x * blockDim.x + threadIdx.x;
  if (idx >= (size_t)k * c) return;
  int i = (int)(idx / c), j = (int)(idx % c);
  xo[(size_t)perm[i] * c + j] += xc[idx];
}

__global__ void k_gather_A(const float* __restrict__ Ain, int nin, const int* __restrict__ perm,
                           float* __restrict__ Aout, int k) {
  size_t idx = (size_t)blockIdx.x * blockDim.x + threadIdx.x;
  if (idx >= (size_t)k * k) return;
  int i = (int)(idx / k), j = (int)(idx % k);
  Aout[idx] = Ain[(size_t)perm[i] * nin + perm[j]];
}

__global__ void k_gather_x(const float* __restrict__ xin, const float* __restrict__ score,
                           const int* __restrict__ perm, float* __restrict__ xout, int k, int c) {
  size_t idx = (size_t)blockIdx.x * blockDim.x + threadIdx.x;
  if (idx >= (size_t)k * c) return;
  int i = (int)(idx / c), j = (int)(idx % c);
  int p = perm[i];
  xout[idx] = xin[(size_t)p * c + j] * score[p];
}

__global__ void k_pnorm(const float* __restrict__ p, float* __restrict__ out, int c) {
  __shared__ float sm[256];
  float a = 0.f;
  for (int j = threadIdx.x; j < c; j += 256) a += p[j] * p[j];
  sm[threadIdx.x] = a; __syncthreads();
  for (int s = 128; s > 0; s >>= 1) {
    if ((int)threadIdx.x < s) sm[threadIdx.x] += sm[threadIdx.x + s];
    __syncthreads();
  }
  if (threadIdx.x == 0) out[0] = sqrtf(sm[0]);
}

// score[i] = tanh( dot(x[i],p) / ||p|| ); one wave32 per node
__global__ void k_scores(const float* __restrict__ x, const float* __restrict__ p,
                         const float* __restrict__ pnorm, float* __restrict__ score,
                         int n, int c) {
  int wv   = blockIdx.x * (blockDim.x >> 5) + (threadIdx.x >> 5);
  int lane = threadIdx.x & 31;
  if (wv >= n) return;
  float acc = 0.f;
  for (int j = lane; j < c; j += 32) acc += x[(size_t)wv * c + j] * p[j];
  for (int o = 16; o > 0; o >>= 1) acc += __shfl_xor(acc, o, 32);
  if (lane == 0) score[wv] = tanhf(acc / pnorm[0]);
}

// single-block shared-memory bitonic sort (descending), keep top-k indices
__global__ __launch_bounds__(1024) void k_topk(const float* __restrict__ score,
                                               int* __restrict__ perm, int n, int kkeep) {
  const int LEN = 4096;
  __shared__ float s[4096];
  __shared__ int   id[4096];
  int t = threadIdx.x;
  for (int i = t; i < LEN; i += 1024) {
    s[i]  = (i < n) ? score[i] : -3.0e38f;
    id[i] = i;
  }
  __syncthreads();
  for (int ksz = 2; ksz <= LEN; ksz <<= 1) {
    for (int j = ksz >> 1; j > 0; j >>= 1) {
      for (int i = t; i < LEN; i += 1024) {
        int ixj = i ^ j;
        if (ixj > i) {
          bool desc = ((i & ksz) == 0);
          float si = s[i], sj = s[ixj];
          bool sw = desc ? (si < sj) : (si > sj);
          if (sw) {
            s[i] = sj; s[ixj] = si;
            int tmp = id[i]; id[i] = id[ixj]; id[ixj] = tmp;
          }
        }
      }
      __syncthreads();
    }
  }
  for (int i = t; i < kkeep; i += 1024) perm[i] = id[i];
}

__global__ void k_colmean(const float* __restrict__ x, float* __restrict__ out, int n, int c) {
  int j = blockIdx.x * blockDim.x + threadIdx.x;
  if (j >= c) return;
  float s = 0.f;
  for (int i = 0; i < n; ++i) s += x[(size_t)i * c + j];
  out[j] = s / (float)n;
}

// ---------------------------------------------------------------------------
// Host orchestration
// ---------------------------------------------------------------------------
extern "C" void kernel_launch(void* const* d_in, const int* in_sizes, int n_in,
                              void* d_out, int out_size, void* d_ws, size_t ws_size,
                              hipStream_t stream) {
  (void)n_in; (void)out_size; (void)ws_size;
  const float* x_in    = (const float*)d_in[0];
  const int*   ei      = (const int*)d_in[1];
  const int    E       = in_sizes[1] / 2;          // edge_index is (2, E)
  const float* w_down0 = (const float*)d_in[3];
  const float* b_down0 = (const float*)d_in[4];
  const float* w_down  = (const float*)d_in[5];
  const float* b_down  = (const float*)d_in[6];
  const float* p_attn  = (const float*)d_in[7];
  const float* w_up    = (const float*)d_in[8];
  const float* b_up    = (const float*)d_in[9];

  // ---- workspace carving (~260 MB total) ----
  char* wbase = (char*)d_ws;
  size_t off = 0;
  auto alloc = [&](size_t bytes) -> void* {
    off = (off + 255) & ~(size_t)255;
    void* p = wbase + off;
    off += bytes;
    return p;
  };
  float*    A0   = (float*)alloc((size_t)4096 * 4096 * 4);
  float*    A1   = (float*)alloc((size_t)2048 * 2048 * 4);
  float*    A2   = (float*)alloc((size_t)1024 * 1024 * 4);
  float*    A3   = (float*)alloc((size_t)512  * 512  * 4);
  float*    Atmp = (float*)alloc((size_t)4096 * 4096 * 4);
  _Float16* H0   = (_Float16*)alloc((size_t)4096 * 4096 * 2);
  _Float16* H1   = (_Float16*)alloc((size_t)4096 * 4096 * 2);
  float*    x0   = (float*)alloc((size_t)4096 * CH * 4);
  float*    x1   = (float*)alloc((size_t)2048 * CH * 4);
  float*    x2   = (float*)alloc((size_t)1024 * CH * 4);
  float*    xa   = (float*)alloc((size_t)4096 * CH * 4);
  float*    xb   = (float*)alloc((size_t)4096 * CH * 4);
  float*    xinb = (float*)alloc((size_t)4096 * CH * 4);
  float*    ybuf = (float*)alloc((size_t)4096 * CH * 4);
  float*    zbuf = (float*)alloc((size_t)4096 * CH * 4);
  _Float16* ytH  = (_Float16*)alloc((size_t)CH * 4096 * 2);
  _Float16* xH   = (_Float16*)alloc((size_t)4096 * CH * 2);
  _Float16* WtH  = (_Float16*)alloc((size_t)CH * CH * 2);
  float*    dinv = (float*)alloc(4096 * 4);
  float*    scor = (float*)alloc(4096 * 4);
  float*    pn   = (float*)alloc(256);
  int*      pm0  = (int*)alloc(2048 * 4);
  int*      pm1  = (int*)alloc(1024 * 4);
  int*      pm2  = (int*)alloc(512 * 4);

  auto blocks = [](size_t tot, int bs) { return dim3((unsigned)cdiv_sz(tot, bs)); };

  auto gemm = [&](const _Float16* Ah, int lda, const _Float16* Bt, int ldb,
                  float* C, int ldc, int M, int N, int K) {
    dim3 grid(N / 64, M / 256);                 // M%256==0, N%64==0 throughout
    k_wmma_gemm<<<grid, dim3(256), 0, stream>>>(Ah, lda, Bt, ldb, C, ldc, M, N, K);
  };

  // GCN layer: x_out = (relu?)( D^-1/2 Ahat^T D^-1/2 (x_in @ W) + b )
  auto gcn = [&](const float* A, int n, const float* xi, int cin,
                 const float* W, const float* b, float* xo, int relu) {
    size_t nx = (size_t)n * cin;
    k_tohalf<<<blocks(nx, 256), 256, 0, stream>>>(xi, xH, nx);
    k_transpose_half<<<blocks((size_t)cin * CH, 256), 256, 0, stream>>>(W, nullptr, WtH, cin, CH);
    gemm(xH, cin, WtH, cin, ybuf, CH, n, CH, cin);                 // y = x@W
    k_dinv<<<blocks(n, 256), 256, 0, stream>>>(A, dinv, n);
    k_transpose_half<<<blocks((size_t)n * CH, 256), 256, 0, stream>>>(ybuf, dinv, ytH, n, CH);
    k_ahatT_half<<<blocks((size_t)n * n, 256), 256, 0, stream>>>(A, H0, n);
    gemm(H0, n, ytH, n, zbuf, CH, n, CH, n);                       // z = Ahat^T (dinv*y)
    k_gcn_epi<<<blocks((size_t)n * CH, 256), 256, 0, stream>>>(zbuf, dinv, b, xo, n, CH, relu);
  };

  // Aout = (A+I)@(A+I) with zero diagonal
  auto augment = [&](const float* A, int n, float* Aout) {
    size_t nn = (size_t)n * n;
    k_aug_half<<<blocks(nn, 256), 256, 0, stream>>>(A, H0, n, 0);
    k_aug_half<<<blocks(nn, 256), 256, 0, stream>>>(A, H1, n, 1);
    gemm(H0, n, H1, n, Aout, n, n, n, n);
    k_zero_diag<<<blocks(n, 256), 256, 0, stream>>>(Aout, n);
  };

  // ---- build adjacency ----
  k_zero_f32<<<dim3(2048), 256, 0, stream>>>(A0, (size_t)4096 * 4096);
  k_build_adj<<<blocks(E, 256), 256, 0, stream>>>(ei, ei + E, A0, E, 4096);

  // ---- encoder ----
  gcn(A0, 4096, x_in, CIN, w_down0, b_down0, x0, 1);

  float* As_[4]  = {A0, A1, A2, A3};
  float* xs_[3]  = {x0, x1, x2};
  int*   pms[3]  = {pm0, pm1, pm2};
  const float* xcur = x0;
  int n = 4096;
  for (int i = 0; i < DEPTHN; ++i) {
    augment(As_[i], n, Atmp);
    int kk = n / 2;
    const float* p = p_attn + i * CH;
    k_pnorm<<<dim3(1), 256, 0, stream>>>(p, pn, CH);
    k_scores<<<dim3((unsigned)((n + 7) / 8)), 256, 0, stream>>>(xcur, p, pn, scor, n, CH);
    k_topk<<<dim3(1), 1024, 0, stream>>>(scor, pms[i], n, kk);
    k_gather_A<<<blocks((size_t)kk * kk, 256), 256, 0, stream>>>(Atmp, n, pms[i], As_[i + 1], kk);
    k_gather_x<<<blocks((size_t)kk * CH, 256), 256, 0, stream>>>(xcur, scor, pms[i], xb, kk, CH);
    float* xdst = (i < DEPTHN - 1) ? xs_[i + 1] : xa;
    gcn(As_[i + 1], kk, xb, CH, w_down + (size_t)i * CH * CH, b_down + i * CH, xdst, 1);
    xcur = xdst;
    n = kk;
  }

  // ---- decoder ----
  for (int i = 0; i < DEPTHN; ++i) {
    int j = DEPTHN - 1 - i;
    int nres = 4096 >> j;              // 1024, 2048, 4096
    int kcur = nres / 2;               // rows of xcur / length of perms[j]
    k_copy_f32<<<blocks((size_t)nres * CH, 256), 256, 0, stream>>>(xs_[j], xinb, (size_t)nres * CH);
    k_scatter_add<<<blocks((size_t)kcur * CH, 256), 256, 0, stream>>>(pms[j], xcur, xinb, kcur, CH);
    float* xo = (xcur == xa) ? xb : xa;
    gcn(As_[j], nres, xinb, CH, w_up + (size_t)i * CH * CH, b_up + i * CH, xo, (i < DEPTHN - 1) ? 1 : 0);
    xcur = xo;
  }

  // ---- global mean pool -> d_out (256 f32) ----
  k_colmean<<<dim3(1), 256, 0, stream>>>(xcur, (float*)d_out, 4096, CH);
}
